// AHGConv_89163521065156
// MI455X (gfx1250) — matrix operand order repaired and checked by reading, not compiled, not run on hardware
//
#include <hip/hip_runtime.h>
#include <hip/hip_bf16.h>

// ---------------------------------------------------------------------------
// AHGConv forward on MI455X (gfx1250, wave32, WMMA).
// All dense GEMMs run through v_wmma_f32_16x16x32_f16 (f16 in, fp32 accum).
// BatchNorm (training mode) is two-pass: GEMM epilogue accumulates per-column
// sum/sumsq (LDS atomics -> global atomics), bn_params builds scale/shift,
// bn_apply does relu(z*a+c) into f16 (next layer) or fp32 (outputs).
// ---------------------------------------------------------------------------

typedef _Float16 h16v __attribute__((ext_vector_type(16)));
typedef _Float16 h8v  __attribute__((ext_vector_type(8)));
typedef float    f8v  __attribute__((ext_vector_type(8)));

namespace {
constexpr int kB    = 64;
constexpr int kNmax = 256;
constexpr int kF    = 128;
constexpr int kH    = 128;
constexpr int kN    = kB * kNmax;   // 16384
constexpr int kE    = kB * 4096;    // 262144
constexpr int kC1   = 64;
constexpr float kEps = 1e-5f;
}

// ---------------------------------------------------------------------------
// Core WMMA GEMM: Z[M,128] = A[M,K] @ W[K,128] + bias, f16 A / f16 Wt(col-major),
// fp32 Z.  Also accumulates per-column sum & sum-of-squares for BatchNorm.
// Block = 256 threads = 8 waves; wave w computes rows 16w..16w+15 of a 128-row
// tile, all 128 columns (8 WMMA 16x16 tiles), K-loop step 32.
//
// Fragment layouts as loaded here (contiguous k, wave32):
//   A 16x32 f16 : lane<16 -> row=lane, 16 consecutive K [k..k+15];
//                 lane>=16 -> row=lane-16, 16 consecutive K [k+16..k+31]
//   B 32x16 f16 : col=lane&15, 16 consecutive K at k+16*(lane>>4)
//                 (so W is stored transposed: Wt[col][k], contiguous in k)
//   C/D 16x16 f32: vgpr r -> row = r + 8*(lane>>4), col = lane&15
// ---------------------------------------------------------------------------
__global__ __launch_bounds__(256) void gemm_bn_stats(
    const _Float16* __restrict__ A, int K,
    const _Float16* __restrict__ Wt,     // [128][K] (transposed weight, f16)
    const float* __restrict__ bias,      // [128]
    float* __restrict__ Z,               // [M][128]
    float* __restrict__ gsum, float* __restrict__ gsq)
{
    __shared__ float ssum[128];
    __shared__ float ssq[128];
    const int tid = threadIdx.x;
    if (tid < 128) { ssum[tid] = 0.f; ssq[tid] = 0.f; }
    __syncthreads();

    const int lane    = tid & 31;
    const int wv      = tid >> 5;
    const int rowBase = blockIdx.x * 128 + wv * 16;
    const int m       = rowBase + (lane & 15);
    const int khalf   = (lane >> 4) * 8;
    const int kb16    = (lane >> 4) * 16;
    const int col     = lane & 15;

    f8v acc[8];
#pragma unroll
    for (int i = 0; i < 8; ++i) acc[i] = (f8v)0.0f;

    const _Float16* Arow = A + (size_t)m * K;
    for (int k0 = 0; k0 < K; k0 += 32) {
        h8v alo = *(const h8v*)(Arow + k0 + kb16);
        h8v ahi = *(const h8v*)(Arow + k0 + kb16 + 8);
        h16v a;
#pragma unroll
        for (int i = 0; i < 8; ++i) { a[i] = alo[i]; a[8 + i] = ahi[i]; }
#pragma unroll
        for (int nt = 0; nt < 8; ++nt) {
            h16v b = *(const h16v*)(Wt + (size_t)(nt * 16 + col) * K + k0 + kb16);
            acc[nt] = __builtin_amdgcn_wmma_f32_16x16x32_f16(
                false, a, false, b, (short)0, acc[nt], false, false);
        }
    }

    const int hi8 = (lane >> 4) * 8;
#pragma unroll
    for (int nt = 0; nt < 8; ++nt) {
        const int c128 = nt * 16 + col;
        const float bv = bias[c128];
        float s = 0.f, q = 0.f;
#pragma unroll
        for (int r = 0; r < 8; ++r) {
            const int row = rowBase + r + hi8;
            const float z = acc[nt][r] + bv;
            Z[(size_t)row * 128 + c128] = z;
            s += z;
            q += z * z;
        }
        atomicAdd(&ssum[c128], s);
        atomicAdd(&ssq[c128], q);
    }
    __syncthreads();
    if (tid < 128) {
        atomicAdd(&gsum[tid], ssum[tid]);
        atomicAdd(&gsq[tid], ssq[tid]);
    }
}

// ---------------------------------------------------------------------------
// Message GEMM: msg = relu(concat(x[src], e) @ msg_W + msg_b), fused with the
// segment-sum scatter (global_atomic_add_f32 into agg[dst]).  K = 256; the
// first 128 of K gathers f16 x rows by src[j], the second 128 converts fp32 e
// rows on the fly.  No BN here (matches reference).
// ---------------------------------------------------------------------------
__global__ __launch_bounds__(256) void gemm_msg_scatter(
    const _Float16* __restrict__ x16,    // [N][128] f16
    const float* __restrict__ ev,        // [E][128] f32
    const int* __restrict__ srcI, const int* __restrict__ dstI,
    const _Float16* __restrict__ Wt,     // [128][256]
    const float* __restrict__ bias,      // [128]
    float* __restrict__ agg)             // [N][128] fp32, pre-zeroed
{
    const int tid     = threadIdx.x;
    const int lane    = tid & 31;
    const int wv      = tid >> 5;
    const int rowBase = blockIdx.x * 128 + wv * 16;
    const int j       = rowBase + (lane & 15);
    const int s       = srcI[j];
    const int khalf   = (lane >> 4) * 8;
    const int kb16    = (lane >> 4) * 16;
    const int col     = lane & 15;

    f8v acc[8];
#pragma unroll
    for (int i = 0; i < 8; ++i) acc[i] = (f8v)0.0f;

    const _Float16* xr = x16 + (size_t)s * 128;
    const float*    er = ev + (size_t)j * 128;

    for (int k0 = 0; k0 < 256; k0 += 32) {
        h16v a;
        if (k0 < 128) {                       // uniform branch: EXEC stays full
            h8v alo = *(const h8v*)(xr + k0 + kb16);
            h8v ahi = *(const h8v*)(xr + k0 + kb16 + 8);
#pragma unroll
            for (int i = 0; i < 8; ++i) { a[i] = alo[i]; a[8 + i] = ahi[i]; }
        } else {
            const int ke = k0 - 128;
#pragma unroll
            for (int i = 0; i < 8; ++i) {
                a[i]     = (_Float16)er[ke + kb16 + i];
                a[8 + i] = (_Float16)er[ke + kb16 + 8 + i];
            }
        }
#pragma unroll
        for (int nt = 0; nt < 8; ++nt) {
            h16v b = *(const h16v*)(Wt + (size_t)(nt * 16 + col) * 256 + k0 + kb16);
            acc[nt] = __builtin_amdgcn_wmma_f32_16x16x32_f16(
                false, a, false, b, (short)0, acc[nt], false, false);
        }
    }

    const int hi8 = (lane >> 4) * 8;
    int dd[8];
#pragma unroll
    for (int r = 0; r < 8; ++r) dd[r] = dstI[rowBase + r + hi8];
#pragma unroll
    for (int nt = 0; nt < 8; ++nt) {
        const int c128 = nt * 16 + col;
        const float bv = bias[c128];
#pragma unroll
        for (int r = 0; r < 8; ++r) {
            const float v = fmaxf(acc[nt][r] + bv, 0.f);
            atomicAdd(&agg[(size_t)dd[r] * 128 + c128], v);
        }
    }
}

// ---- BatchNorm helpers ----------------------------------------------------
__global__ void bn_params(const float* __restrict__ gsum, float* __restrict__ gsq,
                          const float* __restrict__ g, const float* __restrict__ be,
                          float invM, float* __restrict__ a_out, float* __restrict__ c_out,
                          float* __restrict__ gsum_mut)
{
    const int c = threadIdx.x;                 // 128 threads
    const float m = gsum[c] * invM;
    const float v = gsq[c] * invM - m * m;
    const float a = g[c] * rsqrtf(v + kEps);
    a_out[c] = a;
    c_out[c] = be[c] - m * a;
    gsum_mut[c] = 0.f;                         // reset for next GEMM
    gsq[c] = 0.f;
}

__global__ __launch_bounds__(256) void bn_apply(
    const float* __restrict__ Z, const float* __restrict__ a, const float* __restrict__ c,
    _Float16* __restrict__ out16, int ostride, int ocol,
    float* __restrict__ out32)
{
    const size_t i = (size_t)blockIdx.x * 256 + threadIdx.x;
    const int col = (int)(i & 127);
    const size_t row = i >> 7;
    const float y = fmaxf(Z[i] * a[col] + c[col], 0.f);
    if (out16) out16[row * (size_t)ostride + ocol + col] = (_Float16)y;
    if (out32) out32[i] = y;
}

// ---- Element-wise / small helpers -----------------------------------------
__global__ void cvt_f16(const float* __restrict__ in, _Float16* __restrict__ o)
{
    const size_t i = (size_t)blockIdx.x * 256 + threadIdx.x;
    o[i] = (_Float16)in[i];
}

__global__ void add2_f16(const float* __restrict__ a, const float* __restrict__ b,
                         _Float16* __restrict__ o)
{
    const size_t i = (size_t)blockIdx.x * 256 + threadIdx.x;
    o[i] = (_Float16)(a[i] + b[i]);
}

__global__ void edge_sum_f16(const float* __restrict__ nodes, const float* __restrict__ ev,
                             const int* __restrict__ srcI, const int* __restrict__ dstI,
                             _Float16* __restrict__ o)
{
    const size_t i = (size_t)blockIdx.x * 256 + threadIdx.x;
    const size_t j = i >> 7;
    const int f = (int)(i & 127);
    const int s = srcI[j], d = dstI[j];
    o[i] = (_Float16)(nodes[(size_t)d * 128 + f] + nodes[(size_t)s * 128 + f] + ev[i]);
}

// W [K,128] f32 -> Wt [128,K] f16 (so B fragments are contiguous 16-half reads)
__global__ void transpose_w(const float* __restrict__ W, _Float16* __restrict__ Wt, int K)
{
    const int gid = blockIdx.x * 256 + threadIdx.x;   // < K*128
    const int k = gid >> 7;
    const int n = gid & 127;
    Wt[(size_t)n * K + k] = (_Float16)W[gid];
}

// Naive batched fp32 matmul for the tiny pooling matmuls (<=0.7 GFLOP total):
// out[b,m,n] = sum_k A[b,m,k] * W[b,k,n], n in [0,128), batch = 64.
__global__ void bmm_f32(const float* __restrict__ A, const float* __restrict__ W,
                        float* __restrict__ out, int M, int K)
{
    const int gid = blockIdx.x * 256 + threadIdx.x;   // 64*M*128 total
    const int n = gid & 127;
    const int t = gid >> 7;            // b*M + m
    const int b = t / M;
    const int m = t - b * M;
    const float* Ar = A + ((size_t)b * M + m) * K;
    const float* Wb = W + (size_t)b * K * 128;
    float s = 0.f;
    for (int k = 0; k < K; ++k) s = fmaf(Ar[k], Wb[(size_t)k * 128 + n], s);
    out[(size_t)gid] = s;
}

// ---------------------------------------------------------------------------
extern "C" void kernel_launch(void* const* d_in, const int* in_sizes, int n_in,
                              void* d_out, int out_size, void* d_ws, size_t ws_size,
                              hipStream_t stream)
{
    (void)in_sizes; (void)n_in; (void)out_size; (void)ws_size;

    const float* x    = (const float*)d_in[0];
    const float* e    = (const float*)d_in[1];
    const int*   ei   = (const int*)d_in[2];
    const float* attr1= (const float*)d_in[3];
    const float* attr2= (const float*)d_in[4];
    const float* adj1 = (const float*)d_in[5];
    const float* nc1  = (const float*)d_in[6];
    const float* nc2  = (const float*)d_in[7];
    const float* cn1  = (const float*)d_in[8];
    const float* msgW = (const float*)d_in[9];
    const float* msgb = (const float*)d_in[10];
    const float* uW1  = (const float*)d_in[11];
    const float* ub1  = (const float*)d_in[12];
    const float* ug1  = (const float*)d_in[13];
    const float* ube1 = (const float*)d_in[14];
    const float* uW2  = (const float*)d_in[15];
    const float* ub2  = (const float*)d_in[16];
    const float* ug2  = (const float*)d_in[17];
    const float* ube2 = (const float*)d_in[18];
    const float* nW   = (const float*)d_in[19];
    const float* nb   = (const float*)d_in[20];
    const float* ng   = (const float*)d_in[21];
    const float* nbe  = (const float*)d_in[22];
    const float* eW   = (const float*)d_in[23];
    const float* eb   = (const float*)d_in[24];
    const float* eg   = (const float*)d_in[25];
    const float* ebe  = (const float*)d_in[26];
    const float* sW   = (const float*)d_in[27];
    const float* sb   = (const float*)d_in[28];
    const float* sg   = (const float*)d_in[29];
    const float* sbe  = (const float*)d_in[30];

    const int* srcI = ei;          // edge_index[0] (j)
    const int* dstI = ei + kE;     // edge_index[1] (i)

    float* outF = (float*)d_out;
    const size_t XOUT = 0;
    const size_t EOUT = (size_t)kN * kH;                 // 2,097,152
    const size_t SOUT = EOUT + (size_t)kE * kH;          // 35,651,584

    // fp32 z scratch for the two-pass BN lives in d_out's e_out region
    // (E*128 floats; written last, in place, by the final EConv combine).
    float* Z = outF + EOUT;

    // ---- workspace carve-up (~240 MB) ----
    char* w = (char*)d_ws;
    size_t off = 0;
    auto take = [&](size_t bytes) -> void* {
        void* p = (void*)(w + off);
        off += (bytes + 255) & ~(size_t)255;
        return p;
    };
    _Float16* msgWt = (_Float16*)take((size_t)128 * 256 * 2);
    _Float16* uW1t  = (_Float16*)take((size_t)7 * 128 * 128 * 2);
    _Float16* uW2t  = (_Float16*)take((size_t)7 * 128 * 128 * 2);
    _Float16* nWt   = (_Float16*)take((size_t)128 * 256 * 2);
    _Float16* eWt   = (_Float16*)take((size_t)128 * 256 * 2);
    _Float16* sWt   = (_Float16*)take((size_t)128 * 384 * 2);
    float* gstat = (float*)take(1024);      // [sum(128) | sumsq(128)]
    float* gsum = gstat, *gsq = gstat + 128;
    float* bnac = (float*)take(1024);       // [a(128) | c(128)]
    float* bna = bnac, *bnc = bnac + 128;
    _Float16* x16   = (_Float16*)take((size_t)kN * kF * 2);
    float*    up    = (float*)take((size_t)kN * kF * 4);
    float*    aggN  = (float*)take((size_t)kN * kF * 4);
    _Float16* nodeA = (_Float16*)take((size_t)kN * kF * 2);
    _Float16* catN  = (_Float16*)take((size_t)kN * 256 * 2);
    float*    aggS  = (float*)take((size_t)4096 * 128 * 4);
    _Float16* subA  = (_Float16*)take((size_t)4096 * 128 * 2);
    _Float16* catS  = (_Float16*)take((size_t)4096 * 384 * 2);
    _Float16* aeBuf = (_Float16*)take((size_t)kE * 128 * 2);
    _Float16* catE  = (_Float16*)take((size_t)kE * 256 * 2);

    hipMemsetAsync(aggN, 0, (size_t)kN * kF * 4, stream);
    hipMemsetAsync(gstat, 0, 1024, stream);

    // ---- weight prep: fp32 [K,128] -> f16 [128,K] ----
    transpose_w<<<(256 * 128) / 256, 256, 0, stream>>>(msgW, msgWt, 256);
    for (int i = 0; i < 7; ++i) {
        transpose_w<<<(128 * 128) / 256, 256, 0, stream>>>(uW1 + (size_t)i * 16384, uW1t + (size_t)i * 16384, 128);
        transpose_w<<<(128 * 128) / 256, 256, 0, stream>>>(uW2 + (size_t)i * 16384, uW2t + (size_t)i * 16384, 128);
    }
    transpose_w<<<(256 * 128) / 256, 256, 0, stream>>>(nW, nWt, 256);
    transpose_w<<<(256 * 128) / 256, 256, 0, stream>>>(eW, eWt, 256);
    transpose_w<<<(384 * 128) / 256, 256, 0, stream>>>(sW, sWt, 384);

    cvt_f16<<<(kN * kF) / 256, 256, 0, stream>>>(x, x16);

    // up_nodes = nc1 @ attr1   [B,256,64]x[B,64,128] -> [N,128] fp32
    bmm_f32<<<(kB * kNmax * 128) / 256, 256, 0, stream>>>(nc1, attr1, up, kNmax, 64);

    // GEMM + BN + relu helper (two-pass BN through Z / gstat)
    auto upd_gemm = [&](const _Float16* A, int M, int K, const _Float16* Wt,
                        const float* bias, const float* g, const float* be,
                        _Float16* o16, int ostride, int ocol, float* o32) {
        gemm_bn_stats<<<M / 128, 256, 0, stream>>>(A, K, Wt, bias, Z, gsum, gsq);
        bn_params<<<1, 128, 0, stream>>>(gsum, gsq, g, be, 1.0f / (float)M, bna, bnc, gsum);
        bn_apply<<<(unsigned)(((size_t)M * 128) / 256), 256, 0, stream>>>(Z, bna, bnc, o16, ostride, ocol, o32);
    };

    // =================== NConv ===================
    // msg = relu([x[src]||e] @ msg_W + msg_b); aggN[dst] += msg
    gemm_msg_scatter<<<kE / 128, 256, 0, stream>>>(x16, e, srcI, dstI, msgWt, msgb, aggN);
    // branch 0: upd(aggN + x, 0)
    add2_f16<<<(kN * kF) / 256, 256, 0, stream>>>(aggN, x, nodeA);
    upd_gemm(nodeA, kN, 128, uW1t, ub1, ug1, ube1, nodeA, 128, 0, nullptr);
    upd_gemm(nodeA, kN, 128, uW2t, ub2, ug2, ube2, catN, 256, 0, nullptr);
    // branch 1: upd(up + x, 1)
    add2_f16<<<(kN * kF) / 256, 256, 0, stream>>>(up, x, nodeA);
    upd_gemm(nodeA, kN, 128, uW1t + 16384, ub1 + 128, ug1 + 128, ube1 + 128, nodeA, 128, 0, nullptr);
    upd_gemm(nodeA, kN, 128, uW2t + 16384, ub2 + 128, ug2 + 128, ube2 + 128, catN, 256, 128, nullptr);
    // combine -> x_out
    upd_gemm(catN, kN, 256, nWt, nb, ng, nbe, nullptr, 0, 0, outF + XOUT);

    // =================== SubConv =================
    // branch 4: adj1 @ attr1
    bmm_f32<<<(kB * kC1 * 128) / 256, 256, 0, stream>>>(adj1, attr1, aggS, kC1, kC1);
    add2_f16<<<(4096 * 128) / 256, 256, 0, stream>>>(aggS, attr1, subA);
    upd_gemm(subA, 4096, 128, uW1t + (size_t)4 * 16384, ub1 + 4 * 128, ug1 + 4 * 128, ube1 + 4 * 128, subA, 128, 0, nullptr);
    upd_gemm(subA, 4096, 128, uW2t + (size_t)4 * 16384, ub2 + 4 * 128, ug2 + 4 * 128, ube2 + 4 * 128, catS, 384, 0, nullptr);
    // branch 5: nc2 @ attr2
    bmm_f32<<<(kB * kC1 * 128) / 256, 256, 0, stream>>>(nc2, attr2, aggS, kC1, 8);
    add2_f16<<<(4096 * 128) / 256, 256, 0, stream>>>(aggS, attr1, subA);
    upd_gemm(subA, 4096, 128, uW1t + (size_t)5 * 16384, ub1 + 5 * 128, ug1 + 5 * 128, ube1 + 5 * 128, subA, 128, 0, nullptr);
    upd_gemm(subA, 4096, 128, uW2t + (size_t)5 * 16384, ub2 + 5 * 128, ug2 + 5 * 128, ube2 + 5 * 128, catS, 384, 128, nullptr);
    // branch 6: cn1 @ attr0 (attr0 = x reshaped [B,256,128])
    bmm_f32<<<(kB * kC1 * 128) / 256, 256, 0, stream>>>(cn1, x, aggS, kC1, 256);
    add2_f16<<<(4096 * 128) / 256, 256, 0, stream>>>(aggS, attr1, subA);
    upd_gemm(subA, 4096, 128, uW1t + (size_t)6 * 16384, ub1 + 6 * 128, ug1 + 6 * 128, ube1 + 6 * 128, subA, 128, 0, nullptr);
    upd_gemm(subA, 4096, 128, uW2t + (size_t)6 * 16384, ub2 + 6 * 128, ug2 + 6 * 128, ube2 + 6 * 128, catS, 384, 256, nullptr);
    // combine -> s_out
    upd_gemm(catS, 4096, 384, sWt, sb, sg, sbe, nullptr, 0, 0, outF + SOUT);

    // =================== EConv ===================
    // branch 2: upd(x[dst]+x[src]+e, 2)
    edge_sum_f16<<<(unsigned)(((size_t)kE * 128) / 256), 256, 0, stream>>>(x, e, srcI, dstI, aeBuf);
    upd_gemm(aeBuf, kE, 128, uW1t + (size_t)2 * 16384, ub1 + 2 * 128, ug1 + 2 * 128, ube1 + 2 * 128, aeBuf, 128, 0, nullptr);
    upd_gemm(aeBuf, kE, 128, uW2t + (size_t)2 * 16384, ub2 + 2 * 128, ug2 + 2 * 128, ube2 + 2 * 128, catE, 256, 0, nullptr);
    // branch 3: upd(up[dst]+up[src]+e, 3)
    edge_sum_f16<<<(unsigned)(((size_t)kE * 128) / 256), 256, 0, stream>>>(up, e, srcI, dstI, aeBuf);
    upd_gemm(aeBuf, kE, 128, uW1t + (size_t)3 * 16384, ub1 + 3 * 128, ug1 + 3 * 128, ube1 + 3 * 128, aeBuf, 128, 0, nullptr);
    upd_gemm(aeBuf, kE, 128, uW2t + (size_t)3 * 16384, ub2 + 3 * 128, ug2 + 3 * 128, ube2 + 3 * 128, catE, 256, 128, nullptr);
    // combine -> e_out (bn_apply is element-wise in place over Z = e_out region)
    upd_gemm(catE, kE, 256, eWt, eb, eg, ebe, nullptr, 0, 0, outF + EOUT);
}
